// Seq2SeqAttention_7516192768831
// MI455X (gfx1250) — matrix operand order repaired and checked by reading, not compiled, not run on hardware
//
#include <hip/hip_runtime.h>
#include <hip/hip_bf16.h>

typedef __attribute__((ext_vector_type(16))) __bf16       v16bf;
typedef __attribute__((ext_vector_type(8)))  float        v8f;
typedef __attribute__((ext_vector_type(4)))  unsigned int u32x4;
typedef __attribute__((ext_vector_type(4)))  __bf16       v4bf;
typedef int v4i __attribute__((vector_size(4 * sizeof(int))));

#define HS 1024   // HS_ENC == HS_DEC == ATTN
#define TT 2048
#define NB 32

#if defined(__gfx1250__) && __has_builtin(__builtin_amdgcn_global_load_async_to_lds_b128)
#define HAVE_ASYNC 1
#else
#define HAVE_ASYNC 0
#endif

#if HAVE_ASYNC
#define GLOBAL_AS __attribute__((address_space(1)))
#define LDS_AS    __attribute__((address_space(3)))
__device__ __forceinline__ void wait_async0() {
#if __has_builtin(__builtin_amdgcn_s_wait_asynccnt)
    __builtin_amdgcn_s_wait_asynccnt(0);
#else
    asm volatile("s_wait_asynccnt 0x0" ::: "memory");
#endif
}
#endif

// ---------------- prep 1: convert We = W[:,1024:2048] (f32, row stride 2048) -> bf16 ----------------
__global__ void __launch_bounds__(256)
conv_we_kernel(const float* __restrict__ W, __bf16* __restrict__ Web) {
    int idx = blockIdx.x * 256 + threadIdx.x;      // 0 .. 1M-1
    int a = idx >> 10, k = idx & 1023;
    Web[idx] = (__bf16)W[(size_t)a * 2048 + 1024 + k];
}

// ---------------- prep 2: decb[n][a] = b[a] + sum_d h_dec[n][d] * W[a][d] ----------------
__global__ void __launch_bounds__(256)
decproj_kernel(const float* __restrict__ h_dec, const float* __restrict__ W,
               const float* __restrict__ b, float* __restrict__ decb) {
    int idx = blockIdx.x * 256 + threadIdx.x;      // 0 .. 32767
    int n = idx >> 10, a = idx & 1023;
    const float4* hd4 = (const float4*)(h_dec + (size_t)n * HS);
    const float4* wr4 = (const float4*)(W + (size_t)a * 2048);
    float acc = b[a];
    #pragma unroll 4
    for (int d = 0; d < HS / 4; ++d) {
        float4 x = hd4[d], y = wr4[d];
        acc += x.x * y.x + x.y * y.y + x.z * y.z + x.w * y.w;
    }
    decb[idx] = acc;
}

// Stage one 16x1024 bf16 slab of Web into a packed [kk][lane][16] LDS buffer.
// 16 chunks of 16B per thread (128 threads). Async path uses GLOBAL_LOAD_ASYNC_TO_LDS_B128.
__device__ __forceinline__ void
stage_B_slab(const __bf16* __restrict__ Web, __bf16* __restrict__ bbuf, int a_base, int tid) {
    #pragma unroll
    for (int it = 0; it < 16; ++it) {
        int c    = tid + it * 128;
        int row  = c >> 7;            // a_local 0..15
        int col8 = c & 127;
        int k    = col8 * 8;
        int kk   = k >> 5, r5 = k & 31;     // r5 in {0,8,16,24}
        int lane_t = row + 16 * ((r5 >> 3) & 1);
        int elem   = 8 * (r5 >> 4);         // 0 or 8
        const __bf16* gsrc = &Web[(size_t)(a_base + row) * HS + k];
        __bf16*       ldst = &bbuf[((kk * 32 + lane_t) << 4) + elem];
#if HAVE_ASYNC
        __builtin_amdgcn_global_load_async_to_lds_b128(
            (GLOBAL_AS v4i*)gsrc, (LDS_AS v4i*)ldst, 0, 0);
#else
        *(u32x4*)ldst = *(const u32x4*)gsrc;
#endif
    }
}

// ---------------- main: fused enc_proj (bf16 WMMA) + tanh + dot(v) -> score ----------------
// block = 128 threads (4 waves). Covers rows t_base..t_base+63 of batch n.
// Abuf packed layout: [wave][kk][lane][16] bf16  (matches 16-bit A-matrix 16x32 VGPR layout)
// Bbuf packed layout: [buf][kk][lane][16] bf16   (matches B-matrix 32x16 VGPR layout), double-buffered
__global__ void __launch_bounds__(128)
score_wmma_kernel(const float* __restrict__ h_enc, const __bf16* __restrict__ Web,
                  const float* __restrict__ decb, const float* __restrict__ vvec,
                  float* __restrict__ score) {
    __shared__ __bf16 Abuf[4 * 32 * 32 * 16];    // 131072 B
    __shared__ __bf16 Bbuf[2][32 * 32 * 16];     //  65536 B (double buffered)

    const int n      = blockIdx.y;
    const int t_base = blockIdx.x * 64;
    const int tid    = threadIdx.x;
    const int lane   = tid & 31;
    const int wv     = tid >> 5;
    const int laneN  = lane & 15;
    const int hi     = lane >> 4;

    // ---- stage A tile: h_enc[n][t_base..t_base+63][0..1023] f32 -> packed bf16 LDS ----
    {
        const float4* src4 = (const float4*)(h_enc + ((size_t)n * TT + t_base) * HS);
        for (int c = tid; c < 64 * 256; c += 128) {
            int row  = c >> 8;            // 0..63
            int col4 = c & 255;           // float4 index within row
            int k    = col4 * 4;
            int w    = row >> 4, m = row & 15;
            int kk   = k >> 5,  r5 = k & 31;
            int lane_t = m + 16 * ((r5 >> 3) & 1);
            int elem   = (r5 & 7) + 8 * (r5 >> 4);   // 0,4,8,12
            float4 f = src4[(size_t)row * 256 + col4];
            v4bf h;
            h.x = (__bf16)f.x; h.y = (__bf16)f.y; h.z = (__bf16)f.z; h.w = (__bf16)f.w;
            *(v4bf*)&Abuf[(((w * 32 + kk) * 32 + lane_t) << 4) + elem] = h;
        }
    }

    const __bf16* abase = &Abuf[((wv * 32) * 32 + lane) << 4];  // advance 512 elems per kk

    float sAcc[8] = {0.f, 0.f, 0.f, 0.f, 0.f, 0.f, 0.f, 0.f};

    // prologue: stage slab 0 into buffer 0
    stage_B_slab(Web, Bbuf[0], 0, tid);
#if HAVE_ASYNC
    wait_async0();
#endif
    __syncthreads();

    for (int at = 0; at < 64; ++at) {
        const int cur = at & 1;
        // prefetch next slab into the other buffer; latency hides behind the 32 WMMAs below
        if (at < 63)
            stage_B_slab(Web, Bbuf[cur ^ 1], (at + 1) * 16, tid);

        v8f acc = {};
        #pragma unroll
        for (int kk = 0; kk < 32; ++kk) {
            union { v16bf v; u32x4 q[2]; } fa, fb;
            const __bf16* pa = abase + (kk << 9);
            const __bf16* pb = &Bbuf[cur][(kk * 32 + lane) << 4];
            fa.q[0] = *(const u32x4*)(pa);
            fa.q[1] = *(const u32x4*)(pa + 8);
            fb.q[0] = *(const u32x4*)(pb);
            fb.q[1] = *(const u32x4*)(pb + 8);
            acc = __builtin_amdgcn_wmma_f32_16x16x32_bf16(
                    false, fa.v, false, fb.v, (short)0, acc, false, false);
        }

        // D tile: element (M = r + 8*hi, N = laneN); add decb broadcast, tanh, dot with v
        const float db = decb[n * HS + at * 16 + laneN];
        const float vv = vvec[at * 16 + laneN];
        #pragma unroll
        for (int r = 0; r < 8; ++r)
            sAcc[r] += tanhf(acc[r] + db) * vv;

#if HAVE_ASYNC
        wait_async0();      // my async writes into Bbuf[cur^1] have landed
#endif
        __syncthreads();    // all waves' writes visible; all done reading Bbuf[cur]
    }

    // reduce over the 16 N-lanes (within each 16-lane half; bit4 = M+8 half stays put)
    #pragma unroll
    for (int r = 0; r < 8; ++r) {
        float s = sAcc[r];
        s += __shfl_xor(s, 1, 32);
        s += __shfl_xor(s, 2, 32);
        s += __shfl_xor(s, 4, 32);
        s += __shfl_xor(s, 8, 32);
        sAcc[r] = s;
    }
    if (laneN == 0) {
        const int rowbase = t_base + wv * 16 + hi * 8;
        #pragma unroll
        for (int r = 0; r < 8; ++r)
            score[(size_t)n * TT + rowbase + r] = sAcc[r];
    }
}

// ---------------- masked softmax over T per batch row (in place on weights region) ----------------
__global__ void __launch_bounds__(256)
softmax_kernel(float* __restrict__ w, const int* __restrict__ src_lens) {
    __shared__ float redmax[8];
    __shared__ float redsum[8];
    const int n = blockIdx.x;
    const int tid = threadIdx.x;
    const int lane = tid & 31, wv = tid >> 5;
    const int L = src_lens[n];

    float vals[8];
    float mx = -3.4e38f;
    #pragma unroll
    for (int i = 0; i < 8; ++i) {
        int t = i * 256 + tid;
        float s = w[(size_t)n * TT + t];
        s = (t < L) ? s : -1.0e10f;
        vals[i] = s;
        mx = fmaxf(mx, s);
    }
    #pragma unroll
    for (int m = 16; m >= 1; m >>= 1) mx = fmaxf(mx, __shfl_xor(mx, m, 32));
    if (lane == 0) redmax[wv] = mx;
    __syncthreads();
    mx = redmax[0];
    #pragma unroll
    for (int j = 1; j < 8; ++j) mx = fmaxf(mx, redmax[j]);

    float sum = 0.f;
    #pragma unroll
    for (int i = 0; i < 8; ++i) { vals[i] = __expf(vals[i] - mx); sum += vals[i]; }
    #pragma unroll
    for (int m = 16; m >= 1; m >>= 1) sum += __shfl_xor(sum, m, 32);
    if (lane == 0) redsum[wv] = sum;
    __syncthreads();
    sum = 0.f;
    #pragma unroll
    for (int j = 0; j < 8; ++j) sum += redsum[j];

    const float inv = 1.0f / sum;
    #pragma unroll
    for (int i = 0; i < 8; ++i)
        w[(size_t)n * TT + i * 256 + tid] = vals[i] * inv;
}

// ---------------- ctx[n][e] = sum_t weights[n][t] * h_enc[n][t][e] ----------------
__global__ void __launch_bounds__(256)
ctx_kernel(const float* __restrict__ h_enc, const float* __restrict__ w, float* __restrict__ ctx) {
    const int n = blockIdx.y;
    const int e = blockIdx.x * 256 + threadIdx.x;
    const float* he = h_enc + (size_t)n * TT * HS + e;
    const float* wr = w + (size_t)n * TT;
    float acc = 0.f;
    #pragma unroll 8
    for (int t = 0; t < TT; ++t)
        acc += wr[t] * he[(size_t)t * HS];
    ctx[(size_t)n * HS + e] = acc;
}

extern "C" void kernel_launch(void* const* d_in, const int* in_sizes, int n_in,
                              void* d_out, int out_size, void* d_ws, size_t ws_size,
                              hipStream_t stream) {
    const float* h_dec    = (const float*)d_in[0];
    const float* h_enc    = (const float*)d_in[1];
    const int*   src_lens = (const int*)d_in[2];
    const float* W        = (const float*)d_in[3];
    const float* b        = (const float*)d_in[4];
    const float* v        = (const float*)d_in[5];

    float* out     = (float*)d_out;
    float* ctx     = out;                        // 32*1024 f32
    float* weights = out + (size_t)NB * HS;      // 32*2048 f32

    __bf16* Web  = (__bf16*)d_ws;                                            // 2 MB
    float*  decb = (float*)((char*)d_ws + (size_t)HS * HS * sizeof(__bf16)); // 128 KB

    conv_we_kernel <<<(HS * HS) / 256, 256, 0, stream>>>(W, Web);
    decproj_kernel <<<(NB * HS) / 256, 256, 0, stream>>>(h_dec, W, b, decb);
    score_wmma_kernel<<<dim3(TT / 64, NB), 128, 0, stream>>>(h_enc, Web, decb, v, weights);
    softmax_kernel <<<NB, 256, 0, stream>>>(weights, src_lens);
    ctx_kernel     <<<dim3(HS / 256, NB), 256, 0, stream>>>(h_enc, weights, ctx);
}
